// Transformer_33457795236436
// MI455X (gfx1250) — matrix operand order repaired and checked by reading, not compiled, not run on hardware
//
#include <hip/hip_runtime.h>

// ---------------------------------------------------------------------------
// Types for CDNA5 WMMA (wave32): 16x16x32 f16 -> f32 accumulate
// ---------------------------------------------------------------------------
typedef __attribute__((ext_vector_type(16))) _Float16 v16h;
typedef __attribute__((ext_vector_type(8)))  float    v8f;
typedef __attribute__((ext_vector_type(4)))  unsigned int u32x4;
typedef __attribute__((ext_vector_type(8)))  int      i32x8;
typedef __attribute__((ext_vector_type(4)))  int      i32x4;

#define NTOK   21760
#define MENC   43520      // B * NTOK
#define DMODEL 256
#define NQD    300
#define MDEC   600        // B * NQ

// ---------------------------------------------------------------------------
// Tensor Data Mover: DMA a 2D fp32 tile (tile_rows x 32) from global into LDS.
// LDS rows are padded by the TDM (2 DWORDs per 32 -> stride 34 f32, bank-split).
// tensor_dim = remaining rows/cols from the tile origin => hardware zero-fills
// all out-of-bounds elements (handles ragged M/N/K with no guard code).
// ---------------------------------------------------------------------------
__device__ __forceinline__ void tdm_load_2d(
    const float* gptr, int rem_rows, int rem_cols, int ld,
    unsigned lds_off, int tile_rows)
{
  const unsigned long long ga = (unsigned long long)(size_t)gptr;
  const unsigned td0 = (unsigned)(rem_cols > 0 ? rem_cols : 0);  // dim0 = row len
  const unsigned td1 = (unsigned)(rem_rows > 0 ? rem_rows : 0);  // dim1 = rows
  u32x4 g0;
  g0[0] = 1u;                                        // count=1, user descriptor
  g0[1] = lds_off;                                   // LDS byte address
  g0[2] = (unsigned)(ga & 0xffffffffu);              // global addr lo
  g0[3] = (unsigned)((ga >> 32) & 0x1ffffffu) | (2u << 30);  // addr hi | type=2
  i32x8 g1;
  g1[0] = (int)((2u << 16) | (1u << 20) | (4u << 22) | (1u << 25));
          // data_size=4B | pad_enable | pad_interval=32dw | pad_amount=2dw
  g1[1] = (int)((td0 & 0xffffu) << 16);                          // dim0[15:0]
  g1[2] = (int)(((td0 >> 16) & 0xffffu) | ((td1 & 0xffffu) << 16));
  g1[3] = (int)(((td1 >> 16) & 0xffffu) | (32u << 16));          // tile_dim0=32
  g1[4] = (int)(unsigned)tile_rows;                              // tile_dim1
  g1[5] = (int)(unsigned)ld;                                     // dim0 stride
  g1[6] = 0;
  g1[7] = 0;
  i32x4 z4 = {};
#if defined(__clang_major__) && __clang_major__ >= 23
  i32x8 z8 = {};
  __builtin_amdgcn_tensor_load_to_lds(g0, g1, z4, z4, z8, 0);
#else
  __builtin_amdgcn_tensor_load_to_lds(g0, g1, z4, z4, 0);
#endif
}

// ---------------------------------------------------------------------------
// Generic GEMM:  Y[M,N] = act( X[M,K] * W[K,N] + bias ),  fp32 in/out,
// f16 WMMA with f32 accumulation.  TDM-staged, double-buffered LDS tiles.
// Block = 256 threads (8 waves), 64x32 output tile, one wave per 16x16 tile.
// act: 0=none 1=relu 2=sigmoid
// ---------------------------------------------------------------------------
__global__ __launch_bounds__(256) void gemm_wmma(
    const float* __restrict__ X, int ldx,
    const float* __restrict__ W, int ldw,
    const float* __restrict__ bias,
    float* __restrict__ Y, int ldy,
    int M, int N, int K, int act)
{
  __shared__ float As[2][64 * 34];   // [row][k], stride 34 (TDM pad)
  __shared__ float Bs[2][32 * 34];   // [k][n],  stride 34 (TDM pad)

  const int tid  = threadIdx.x;
  const int lane = tid & 31;
  const int wv   = tid >> 5;
  const int half = lane >> 4;
  const int l16  = lane & 15;
  const int blockM = blockIdx.x * 64;
  const int blockN = blockIdx.y * 32;
  const int rowA = (wv & 3) * 16 + l16;   // 0..63 : A row this lane reads
  const int colB = (wv >> 2) * 16 + l16;  // 0..31 : B column this lane reads

  // Opaque store keeps LDS loads from being folded (TDM is the only writer).
  if (M == -1) { As[0][tid] = 0.f; Bs[0][tid] = 0.f; }

  // Prologue: stream first K-slice.
  if (wv == 0)
    tdm_load_2d(X + (size_t)blockM * ldx, M - blockM, K, ldx,
                (unsigned)(size_t)&As[0][0], 64);
  if (wv == 1)
    tdm_load_2d(W + blockN, K, N - blockN, ldw,
                (unsigned)(size_t)&Bs[0][0], 32);

  v8f acc = {};
  for (int k0 = 0; k0 < K; k0 += 32) {
    const int ib = (k0 >> 5) & 1;
    if (wv < 2) __builtin_amdgcn_s_wait_tensorcnt(0);  // tiles[ib] landed
    __syncthreads();

    // Overlap: stream next K-slice into the other buffer while we compute.
    const int k1 = k0 + 32;
    if (k1 < K) {
      if (wv == 0)
        tdm_load_2d(X + (size_t)blockM * ldx + k1, M - blockM, K - k1, ldx,
                    (unsigned)(size_t)&As[ib ^ 1][0], 64);
      if (wv == 1)
        tdm_load_2d(W + (size_t)k1 * ldw + blockN, K - k1, N - blockN, ldw,
                    (unsigned)(size_t)&Bs[ib ^ 1][0], 32);
    }

    // ---- build fragments (f32 LDS -> f16 regs) ----
    v16h a, b;
    const float* ap = &As[ib][0] + rowA * 34;
#pragma unroll
    for (int v = 0; v < 8; ++v) {
      const int kk = ((v >> 2) << 4) + ((v & 3) << 1) + (half << 3);
      const float2 p = *(const float2*)(ap + kk);
      a[2 * v] = (_Float16)p.x; a[2 * v + 1] = (_Float16)p.y;
    }
    const float* bp = &Bs[ib][0] + colB;
#pragma unroll
    for (int j = 0; j < 8; ++j) {
      const int kk = ((j >> 2) << 4) + ((j & 3) << 1) + (half << 3);
      b[2 * j]     = (_Float16)bp[(kk + 0) * 34];
      b[2 * j + 1] = (_Float16)bp[(kk + 1) * 34];
    }
    acc = __builtin_amdgcn_wmma_f32_16x16x32_f16(false, a, false, b,
                                                 (short)0, acc, false, false);
  }

  // ---- epilogue: C/D VGPR r -> M = tileM + r + 8*half, N = tileN + l16 ----
  const int tileM = blockM + (wv & 3) * 16;
  const int tileN = blockN + (wv >> 2) * 16;
  if (blockM + 64 <= M && blockN + 32 <= N) {
    const int col = tileN + l16;
    const float bv = bias ? bias[col] : 0.0f;
#pragma unroll
    for (int r = 0; r < 8; ++r) {
      const int row = tileM + r + half * 8;
      float v = acc[r] + bv;
      if (act == 1)      v = fmaxf(v, 0.0f);
      else if (act == 2) v = 1.0f / (1.0f + __expf(-v));
      Y[(size_t)row * ldy + col] = v;
    }
  } else {
#pragma unroll
    for (int r = 0; r < 8; ++r) {
      const int row = tileM + r + half * 8;
      const int col = tileN + l16;
      if (row < M && col < N) {
        float v = acc[r] + (bias ? bias[col] : 0.0f);
        if (act == 1)      v = fmaxf(v, 0.0f);
        else if (act == 2) v = 1.0f / (1.0f + __expf(-v));
        Y[(size_t)row * ldy + col] = v;
      }
    }
  }
}

// ---------------------------------------------------------------------------
// (B, D, H, W) -> token-major (B, NTOK, D) slice starting at tokStart
// ---------------------------------------------------------------------------
__global__ void to_tokens(const float* __restrict__ in, float* __restrict__ out,
                          int HW, int tokStart)
{
  const size_t idx = (size_t)blockIdx.x * blockDim.x + threadIdx.x;
  const size_t tot = (size_t)2 * HW * DMODEL;
  if (idx >= tot) return;
  const int d  = (int)(idx & 255);
  const size_t rest = idx >> 8;
  const int hw = (int)(rest % HW);
  const int b  = (int)(rest / HW);
  out[((size_t)(b * NTOK + tokStart + hw)) * DMODEL + d] =
      in[((size_t)(b * DMODEL + d)) * HW + hw];
}

__global__ void add_k(const float* __restrict__ a, const float* __restrict__ b,
                      float* __restrict__ c, size_t n)
{
  size_t i = (size_t)blockIdx.x * blockDim.x + threadIdx.x;
  if (i < n) c[i] = a[i] + b[i];
}

__global__ void copy_k(const float* __restrict__ a, float* __restrict__ c, size_t n)
{
  size_t i = (size_t)blockIdx.x * blockDim.x + threadIdx.x;
  if (i < n) c[i] = a[i];
}

// ---------------------------------------------------------------------------
// x = LayerNorm(x + r) over D=256.  One block (256 threads = 8 waves) per row.
// ---------------------------------------------------------------------------
__global__ __launch_bounds__(256) void ln_residual(
    float* __restrict__ x, const float* __restrict__ r,
    const float* __restrict__ g, const float* __restrict__ bt)
{
  const int row = blockIdx.x;
  const int d = threadIdx.x;
  const size_t o = (size_t)row * DMODEL + d;
  const float v = x[o] + r[o];
  __shared__ float red[8];
  float s = v;
#pragma unroll
  for (int k = 16; k; k >>= 1) s += __shfl_xor(s, k, 32);
  if ((d & 31) == 0) red[d >> 5] = s;
  __syncthreads();
  float mu = 0.f;
#pragma unroll
  for (int i = 0; i < 8; ++i) mu += red[i];
  mu *= (1.0f / 256.0f);
  __syncthreads();
  const float dv = v - mu;
  float s2 = dv * dv;
#pragma unroll
  for (int k = 16; k; k >>= 1) s2 += __shfl_xor(s2, k, 32);
  if ((d & 31) == 0) red[d >> 5] = s2;
  __syncthreads();
  float var = 0.f;
#pragma unroll
  for (int i = 0; i < 8; ++i) var += red[i];
  var *= (1.0f / 256.0f);
  x[o] = dv * rsqrtf(var + 1e-5f) * g[d] + bt[d];
}

// ---------------------------------------------------------------------------
// Deformable box attention sampling core. One block per token t, one wave per
// head; lane = channel (HD=32). Softmax over 16 (level,point) logits done
// in-register; bilinear gather from value (B, NTOK, 8, 32).
// ---------------------------------------------------------------------------
__global__ __launch_bounds__(256) void box_sample(
    const float* __restrict__ offs,   // M x 256  (h,l,p,2)
    const float* __restrict__ wtsl,   // M x 128  logits (h, l*4+p)
    const float* __restrict__ value,  // B x NTOK x 256
    const float* __restrict__ ref,    // M x rs
    int rs, int Q, int rot,
    float* __restrict__ out)          // M x 256
{
  const int t = blockIdx.x;
  const int h = threadIdx.x >> 5;
  const int lane = threadIdx.x & 31;
  const int b = t / Q;
  const float* rp = ref + (size_t)t * rs;
  const float cx = rp[0], cy = rp[1], bw = rp[3], bh = rp[4];
  float cs = 1.0f, sn = 0.0f;
  if (rot) { const float ang = rp[6]; cs = cosf(ang); sn = sinf(ang); }

  const float* wp = wtsl + (size_t)t * 128 + h * 16;
  float wexp[16];
  float mx = -1e30f;
#pragma unroll
  for (int i = 0; i < 16; ++i) mx = fmaxf(mx, wp[i]);
  float sum = 0.f;
#pragma unroll
  for (int i = 0; i < 16; ++i) { const float e = __expf(wp[i] - mx); wexp[i] = e; sum += e; }
  const float inv = 1.0f / sum;

  const float* op = offs + (size_t)t * 256 + h * 32;
  const int Hs[4] = {128, 64, 32, 16};
  const int St[4] = {0, 16384, 20480, 21504};
  float acc = 0.f;
#pragma unroll
  for (int l = 0; l < 4; ++l) {
    const int H = Hs[l], W = Hs[l], st = St[l];
#pragma unroll
    for (int p = 0; p < 4; ++p) {
      const float ox = op[l * 8 + p * 2 + 0] * (0.5f * bw);
      const float oy = op[l * 8 + p * 2 + 1] * (0.5f * bh);
      const float rx = rot ? (ox * cs - oy * sn) : ox;
      const float ry = rot ? (ox * sn + oy * cs) : oy;
      const float x = (cx + rx) * W - 0.5f;
      const float y = (cy + ry) * H - 0.5f;
      const float x0 = floorf(x), y0 = floorf(y);
      const float dx = x - x0, dy = y - y0;
      const float wgt = wexp[l * 4 + p] * inv;
#pragma unroll
      for (int c = 0; c < 4; ++c) {
        const int cxi = (int)x0 + (c & 1);
        const int cyi = (int)y0 + (c >> 1);
        const float cw = ((c & 1) ? dx : 1.f - dx) * ((c >> 1) ? dy : 1.f - dy);
        const bool valid = (cxi >= 0) & (cxi < W) & (cyi >= 0) & (cyi < H);
        const int xc = min(max(cxi, 0), W - 1);
        const int yc = min(max(cyi, 0), H - 1);
        const int idx = st + yc * W + xc;
        const float gv = value[((size_t)(b * NTOK + idx)) * 256 + h * 32 + lane];
        acc += wgt * (valid ? cw : 0.f) * gv;
      }
    }
  }
  out[(size_t)t * 256 + h * 32 + lane] = acc;
}

// ---------------------------------------------------------------------------
// Decoder self-attention core: flash-style online softmax.
// One block per query token t (=b*300+q), wave = head, lane = channel.
// ---------------------------------------------------------------------------
__global__ __launch_bounds__(256) void mha_core(
    const float* __restrict__ qkp,  // (B*300) x 512  [qp | kp]
    const float* __restrict__ vp,   // (B*300) x 256
    float* __restrict__ o)          // (B*300) x 256
{
  const int t = blockIdx.x;
  const int h = threadIdx.x >> 5;
  const int lane = threadIdx.x & 31;
  const int base = (t / NQD) * NQD;
  const float qd = qkp[(size_t)t * 512 + h * 32 + lane];
  const float scale = 0.17677669529663687f; // 32^-0.5
  float m = -1e30f, s = 0.f, acc = 0.f;
  for (int k = 0; k < NQD; ++k) {
    float dd = qd * qkp[(size_t)(base + k) * 512 + 256 + h * 32 + lane];
#pragma unroll
    for (int off = 16; off; off >>= 1) dd += __shfl_xor(dd, off, 32);
    dd *= scale;
    const float mn = fmaxf(m, dd);
    const float corr = __expf(m - mn);
    const float pr = __expf(dd - mn);
    const float vd = vp[(size_t)(base + k) * 256 + h * 32 + lane];
    s = s * corr + pr;
    acc = acc * corr + pr * vd;
    m = mn;
  }
  o[(size_t)t * 256 + h * 32 + lane] = acc / s;
}

// ---------------------------------------------------------------------------
// Anchors (also written straight to the output buffer)
// ---------------------------------------------------------------------------
__global__ void anchors_k(float* __restrict__ anch, float* __restrict__ out_anch)
{
  const int t = blockIdx.x * blockDim.x + threadIdx.x;
  if (t >= 2 * NTOK) return;
  const int tok = t % NTOK;
  int st, H, W;
  if (tok < 16384)      { st = 0;     H = 128; W = 128; }
  else if (tok < 20480) { st = 16384; H = 64;  W = 64;  }
  else if (tok < 21504) { st = 20480; H = 32;  W = 32;  }
  else                  { st = 21504; H = 16;  W = 16;  }
  const int r = tok - st;
  const int y = r / W, x = r % W;
  const float vals[7] = { (x + 0.5f) / W, (y + 0.5f) / H, 0.5f, 0.025f, 0.025f, 0.5f, 0.f };
#pragma unroll
  for (int j = 0; j < 7; ++j) {
    anch[(size_t)t * 7 + j] = vals[j];
    out_anch[(size_t)t * 7 + j] = vals[j];
  }
}

// ---------------------------------------------------------------------------
// Iterative top-k (k=300) per batch. Destroys probs. JAX tie-break: lower idx.
// ---------------------------------------------------------------------------
__global__ __launch_bounds__(1024) void topk300(
    float* __restrict__ probs, float* __restrict__ outv, int* __restrict__ outi)
{
  const int b = blockIdx.x;
  float* p = probs + (size_t)b * NTOK;
  __shared__ float sv[1024];
  __shared__ int   si[1024];
  for (int sel = 0; sel < NQD; ++sel) {
    float best = -1e30f; int bi = 0x7fffffff;
    for (int j = threadIdx.x; j < NTOK; j += 1024) {
      const float v = p[j];
      if (v > best || (v == best && j < bi)) { best = v; bi = j; }
    }
    sv[threadIdx.x] = best; si[threadIdx.x] = bi;
    __syncthreads();
    for (int s = 512; s > 0; s >>= 1) {
      if (threadIdx.x < (unsigned)s) {
        const float ov = sv[threadIdx.x + s]; const int oi = si[threadIdx.x + s];
        if (ov > sv[threadIdx.x] || (ov == sv[threadIdx.x] && oi < si[threadIdx.x])) {
          sv[threadIdx.x] = ov; si[threadIdx.x] = oi;
        }
      }
      __syncthreads();
    }
    if (threadIdx.x == 0) {
      outv[b * NQD + sel] = sv[0];
      outi[b * NQD + sel] = si[0];
      p[si[0]] = -1e30f;
    }
    __syncthreads();
  }
}

// Gather selected refs, build refw10, emit init_ref + idx outputs.
__global__ void refsel_k(const float* __restrict__ refall,
                         const float* __restrict__ tv, const int* __restrict__ ti,
                         float* __restrict__ refw, float* __restrict__ out_initref,
                         float* __restrict__ out_idx)
{
  const int t = blockIdx.x * blockDim.x + threadIdx.x;
  if (t >= MDEC) return;
  const int b = t / NQD;
  const int idx = ti[t];
#pragma unroll
  for (int j = 0; j < 7; ++j) {
    const float v = refall[((size_t)(b * NTOK + idx)) * 7 + j];
    refw[(size_t)t * 10 + j] = v;
    out_initref[(size_t)t * 7 + j] = v;
  }
  const float pv = tv[t];
  refw[t * 10 + 7] = pv; refw[t * 10 + 8] = pv; refw[t * 10 + 9] = pv;
  out_idx[t] = (float)idx;
}

// Per-decoder-layer head update: new_ref = refw[:7]+boxd; refw[7:]=sigmoid(cls)
__global__ void det_update_k(const float* __restrict__ boxd,
                             const float* __restrict__ cls,
                             float* __restrict__ refw, float* __restrict__ out_iref)
{
  const int t = blockIdx.x * blockDim.x + threadIdx.x;
  if (t >= MDEC) return;
#pragma unroll
  for (int j = 0; j < 7; ++j) {
    const float nr = refw[t * 10 + j] + boxd[t * 7 + j];
    out_iref[t * 7 + j] = nr;
    refw[t * 10 + j] = nr;
  }
#pragma unroll
  for (int j = 0; j < 3; ++j)
    refw[t * 10 + 7 + j] = 1.f / (1.f + __expf(-cls[t * 3 + j]));
}

// ---------------------------------------------------------------------------
// Host side
// ---------------------------------------------------------------------------
static inline void gemm(hipStream_t s, const float* X, int ldx,
                        const float* W, int ldw, const float* b,
                        float* Y, int ldy, int M, int N, int K, int act)
{
  dim3 g((M + 63) / 64, (N + 31) / 32);
  gemm_wmma<<<g, 256, 0, s>>>(X, ldx, W, ldw, b, Y, ldy, M, N, K, act);
}
static inline void ew_add(hipStream_t s, const float* a, const float* b, float* c, size_t n)
{ add_k<<<(unsigned)((n + 255) / 256), 256, 0, s>>>(a, b, c, n); }
static inline void ew_copy(hipStream_t s, const float* a, float* c, size_t n)
{ copy_k<<<(unsigned)((n + 255) / 256), 256, 0, s>>>(a, c, n); }

// ---- JAX pytree (sorted-key) flat parameter offsets, in floats ----
static const size_t DL_SZ = 1154944;
static const size_t DL_CR_OFFS_B = 0,      DL_CR_OFFS_W = 256;
static const size_t DL_CR_OUT_B  = 65792,  DL_CR_OUT_W  = 66048;
static const size_t DL_CR_VAL_B  = 131584, DL_CR_VAL_W  = 131840;
static const size_t DL_CR_WTS_B  = 197376, DL_CR_WTS_W  = 197504;
static const size_t DL_L1_B = 230272, DL_L1_W = 231296;
static const size_t DL_L2_B = 493440, DL_L2_W = 493696;
static const size_t DL_N1_B = 755840, DL_N1_G = 756096;
static const size_t DL_N2_B = 756352, DL_N2_G = 756608;
static const size_t DL_N3_B = 756864, DL_N3_G = 757120;
static const size_t DL_PM0_B = 757376, DL_PM0_W = 757632;
static const size_t DL_PM1_B = 760192, DL_PM1_W = 760448;
static const size_t DL_PM2_B = 825984, DL_PM2_W = 826240;
static const size_t DL_SELF_INB = 891776, DL_SELF_INW = 892544;
static const size_t DL_SELF_OUT_B = 1089152, DL_SELF_OUT_W = 1089408;
static const size_t DT_SZ = 2570;
static const size_t DT_BOX_B = 0, DT_BOX_W = 7, DT_CLS_B = 1799, DT_CLS_W = 1802;
static const size_t EL_SZ = 756864;
static const size_t EL_OFFS_B = 0,      EL_OFFS_W = 256;
static const size_t EL_OUT_B  = 65792,  EL_OUT_W  = 66048;
static const size_t EL_VAL_B  = 131584, EL_VAL_W  = 131840;
static const size_t EL_WTS_B  = 197376, EL_WTS_W  = 197504;
static const size_t EL_L1_B = 230272, EL_L1_W = 231296;
static const size_t EL_L2_B = 493440, EL_L2_W = 493696;
static const size_t EL_N1_B = 755840, EL_N1_G = 756096;
static const size_t EL_N2_B = 756352, EL_N2_G = 756608;
static const size_t DEC_BASE = 0;
static const size_t DET_BASE = 6929664;
static const size_t ENC_BASE = 6945084;
static const size_t PROP_BASE = 11486268;  // box.b(7), box.w(1792), cls.b(1), cls.w(256)

// ---- output offsets (floats) ----
static const size_t OUT_HS = 0;             // 6*2*300*256
static const size_t OUT_INITREF = 921600;   // 2*300*7
static const size_t OUT_IREFS = 925800;     // 6*2*300*7
static const size_t OUT_MEM = 951000;       // 2*21760*256
static const size_t OUT_ANCH = 12092120;    // 2*21760*7
static const size_t OUT_IDX = 12396760;     // 2*300

// ---- workspace offsets (floats) ----
static const size_t WS_POS    = 0;                 // 11141120
static const size_t WS_MEM    = 11141120;          // 11141120
static const size_t WS_T0     = 22282240;          // src+pos / sampled
static const size_t WS_VAL    = 33423360;          // value projection
static const size_t WS_T1     = 44564480;          // offs / proj / ff2 / boxdelta
static const size_t WS_WTS    = 55705600;          // 5570560
static const size_t WS_FF     = 61276160;          // 44564480
static const size_t WS_ANCH   = 105840640;         // 304640
static const size_t WS_PROBS  = 106145280;         // 43520
static const size_t WS_REFALL = 106188800;         // 304640
static const size_t WS_TOPKV  = 106493440;         // 600
static const size_t WS_TOPKI  = 106494040;         // 600 (int)
static const size_t WS_REFW   = 106494640;         // 6000
static const size_t WS_QPOS   = 106500640;         // 153600
static const size_t WS_G1     = 106654240;         // 153600
static const size_t WS_G2     = 106807840;         // 153600
static const size_t WS_QUERY  = 106961440;         // 153600
static const size_t WS_QK     = 107115040;         // 153600
static const size_t WS_QKP    = 107268640;         // 307200
static const size_t WS_VP     = 107575840;         // 153600
static const size_t WS_O      = 107729440;         // 153600
static const size_t WS_T2     = 107883040;         // 153600
static const size_t WS_F1D    = 108036640;         // 614400
static const size_t WS_CLS    = 108651040;         // 1800
static const size_t WS_BOXD   = 108652840;         // 4200

extern "C" void kernel_launch(void* const* d_in, const int* in_sizes, int n_in,
                              void* d_out, int out_size, void* d_ws, size_t ws_size,
                              hipStream_t stream)
{
  (void)in_sizes; (void)n_in; (void)out_size; (void)ws_size;
  const float* P = (const float*)d_in[8];
  float* out = (float*)d_out;
  float* ws  = (float*)d_ws;

  const size_t NELEM = (size_t)MENC * DMODEL;  // 11,141,120

  // ---- Stage inputs: (B,D,H,W) -> (B,NTOK,D) for src and pos ----
  const int HWs[4] = {16384, 4096, 1024, 256};
  const int Sts[4] = {0, 16384, 20480, 21504};
  for (int lvl = 0; lvl < 4; ++lvl) {
    const size_t tot = (size_t)2 * DMODEL * HWs[lvl];
    const unsigned blocks = (unsigned)((tot + 255) / 256);
    to_tokens<<<blocks, 256, 0, stream>>>((const float*)d_in[2 * lvl],     ws + WS_MEM, HWs[lvl], Sts[lvl]);
    to_tokens<<<blocks, 256, 0, stream>>>((const float*)d_in[2 * lvl + 1], ws + WS_POS, HWs[lvl], Sts[lvl]);
  }
  anchors_k<<<(2 * NTOK + 255) / 256, 256, 0, stream>>>(ws + WS_ANCH, out + OUT_ANCH);

  // ---- Encoder: 6 layers ----
  for (int l = 0; l < 6; ++l) {
    const float* E = P + ENC_BASE + (size_t)l * EL_SZ;
    ew_add(stream, ws + WS_MEM, ws + WS_POS, ws + WS_T0, NELEM);               // src + pos
    gemm(stream, ws + WS_MEM, 256, E + EL_VAL_W, 256, E + EL_VAL_B, ws + WS_VAL, 256, MENC, 256, 256, 0);
    gemm(stream, ws + WS_T0, 256, E + EL_OFFS_W, 256, E + EL_OFFS_B, ws + WS_T1, 256, MENC, 256, 256, 0);
    gemm(stream, ws + WS_T0, 256, E + EL_WTS_W, 128, E + EL_WTS_B, ws + WS_WTS, 128, MENC, 128, 256, 0);
    box_sample<<<MENC, 256, 0, stream>>>(ws + WS_T1, ws + WS_WTS, ws + WS_VAL,
                                         ws + WS_ANCH, 7, NTOK, 0, ws + WS_T0);
    gemm(stream, ws + WS_T0, 256, E + EL_OUT_W, 256, E + EL_OUT_B, ws + WS_T1, 256, MENC, 256, 256, 0);
    ln_residual<<<MENC, 256, 0, stream>>>(ws + WS_MEM, ws + WS_T1, E + EL_N1_G, E + EL_N1_B);
    gemm(stream, ws + WS_MEM, 256, E + EL_L1_W, 1024, E + EL_L1_B, ws + WS_FF, 1024, MENC, 1024, 256, 1);
    gemm(stream, ws + WS_FF, 1024, E + EL_L2_W, 256, E + EL_L2_B, ws + WS_T1, 256, MENC, 256, 1024, 0);
    ln_residual<<<MENC, 256, 0, stream>>>(ws + WS_MEM, ws + WS_T1, E + EL_N2_G, E + EL_N2_B);
  }
  ew_copy(stream, ws + WS_MEM, out + OUT_MEM, NELEM);

  // ---- Proposal heads + top-k selection ----
  const float* PR = P + PROP_BASE;
  gemm(stream, ws + WS_MEM, 256, PR + 1800, 1, PR + 1799, ws + WS_PROBS, 1, MENC, 1, 256, 2); // sigmoid(cls)
  gemm(stream, ws + WS_MEM, 256, PR + 7, 7, PR + 0, ws + WS_T1, 7, MENC, 7, 256, 0);          // box delta
  ew_add(stream, ws + WS_ANCH, ws + WS_T1, ws + WS_REFALL, (size_t)2 * NTOK * 7);
  topk300<<<2, 1024, 0, stream>>>(ws + WS_PROBS, ws + WS_TOPKV, (int*)(ws + WS_TOPKI));
  refsel_k<<<(MDEC + 255) / 256, 256, 0, stream>>>(ws + WS_REFALL, ws + WS_TOPKV,
      (const int*)(ws + WS_TOPKI), ws + WS_REFW, out + OUT_INITREF, out + OUT_IDX);

  // ---- Decoder: 6 layers ----
  for (int i = 0; i < 6; ++i) {
    const float* L = P + DEC_BASE + (size_t)i * DL_SZ;
    const float* T = P + DET_BASE + (size_t)i * DT_SZ;
    // pos_mlp(refw10)
    gemm(stream, ws + WS_REFW, 10, L + DL_PM0_W, 256, L + DL_PM0_B, ws + WS_G1, 256, MDEC, 256, 10, 1);
    gemm(stream, ws + WS_G1, 256, L + DL_PM1_W, 256, L + DL_PM1_B, ws + WS_G2, 256, MDEC, 256, 256, 1);
    gemm(stream, ws + WS_G2, 256, L + DL_PM2_W, 256, L + DL_PM2_B, ws + WS_QPOS, 256, MDEC, 256, 256, 0);
    const float* qkin;
    if (i == 0) {
      ew_copy(stream, ws + WS_QPOS, ws + WS_QUERY, (size_t)MDEC * 256);        // query = pos_mlp out
      qkin = ws + WS_QUERY;
    } else {
      ew_add(stream, ws + WS_QUERY, ws + WS_QPOS, ws + WS_QK, (size_t)MDEC * 256);
      qkin = ws + WS_QK;
    }
    // self attention
    gemm(stream, qkin, 256, L + DL_SELF_INW, 768, L + DL_SELF_INB, ws + WS_QKP, 512, MDEC, 512, 256, 0);
    gemm(stream, ws + WS_QUERY, 256, L + DL_SELF_INW + 512, 768, L + DL_SELF_INB + 512,
         ws + WS_VP, 256, MDEC, 256, 256, 0);
    mha_core<<<MDEC, 256, 0, stream>>>(ws + WS_QKP, ws + WS_VP, ws + WS_O);
    gemm(stream, ws + WS_O, 256, L + DL_SELF_OUT_W, 256, L + DL_SELF_OUT_B, ws + WS_T2, 256, MDEC, 256, 256, 0);
    ln_residual<<<MDEC, 256, 0, stream>>>(ws + WS_QUERY, ws + WS_T2, L + DL_N1_G, L + DL_N1_B);
    // cross (deformable box) attention
    const float* cq;
    if (i == 0) cq = ws + WS_QUERY;
    else { ew_add(stream, ws + WS_QUERY, ws + WS_QPOS, ws + WS_QK, (size_t)MDEC * 256); cq = ws + WS_QK; }
    gemm(stream, ws + WS_MEM, 256, L + DL_CR_VAL_W, 256, L + DL_CR_VAL_B, ws + WS_VAL, 256, MENC, 256, 256, 0);
    gemm(stream, cq, 256, L + DL_CR_OFFS_W, 256, L + DL_CR_OFFS_B, ws + WS_T1, 256, MDEC, 256, 256, 0);
    gemm(stream, cq, 256, L + DL_CR_WTS_W, 128, L + DL_CR_WTS_B, ws + WS_WTS, 128, MDEC, 128, 256, 0);
    box_sample<<<MDEC, 256, 0, stream>>>(ws + WS_T1, ws + WS_WTS, ws + WS_VAL,
                                         ws + WS_REFW, 10, NQD, 1, ws + WS_T0);
    gemm(stream, ws + WS_T0, 256, L + DL_CR_OUT_W, 256, L + DL_CR_OUT_B, ws + WS_T2, 256, MDEC, 256, 256, 0);
    ln_residual<<<MDEC, 256, 0, stream>>>(ws + WS_QUERY, ws + WS_T2, L + DL_N2_G, L + DL_N2_B);
    // FFN
    gemm(stream, ws + WS_QUERY, 256, L + DL_L1_W, 1024, L + DL_L1_B, ws + WS_F1D, 1024, MDEC, 1024, 256, 1);
    gemm(stream, ws + WS_F1D, 1024, L + DL_L2_W, 256, L + DL_L2_B, ws + WS_T2, 256, MDEC, 256, 1024, 0);
    ln_residual<<<MDEC, 256, 0, stream>>>(ws + WS_QUERY, ws + WS_T2, L + DL_N3_G, L + DL_N3_B);
    // detection heads + ref update
    gemm(stream, ws + WS_QUERY, 256, T + DT_CLS_W, 3, T + DT_CLS_B, ws + WS_CLS, 3, MDEC, 3, 256, 0);
    gemm(stream, ws + WS_QUERY, 256, T + DT_BOX_W, 7, T + DT_BOX_B, ws + WS_BOXD, 7, MDEC, 7, 256, 0);
    det_update_k<<<(MDEC + 255) / 256, 256, 0, stream>>>(ws + WS_BOXD, ws + WS_CLS,
        ws + WS_REFW, out + OUT_IREFS + (size_t)i * MDEC * 7);
    ew_copy(stream, ws + WS_QUERY, out + OUT_HS + (size_t)i * MDEC * 256, (size_t)MDEC * 256);
  }
}
